// SAKTWithDecay_81879256531517
// MI455X (gfx1250) — matrix operand order repaired and checked by reading, not compiled, not run on hardware
//
#include <hip/hip_runtime.h>

typedef __attribute__((ext_vector_type(16))) __bf16 bf16x16;
typedef __attribute__((ext_vector_type(8)))  float  f32x8;
typedef unsigned int u32x4 __attribute__((ext_vector_type(4)));
typedef int          i32x4 __attribute__((ext_vector_type(4)));
typedef int          i32x8 __attribute__((ext_vector_type(8)));

union AFrag {
  bf16x16 v;
  __bf16  e[16];
  float4  f4[2];
};

constexpr int BATCH = 32;
constexpr int SEQ   = 1024;
constexpr int DM    = 256;
constexpr int NH    = 8;
constexpr int HDIM  = 32;
constexpr int DFF   = 512;
constexpr int ROWS  = BATCH * SEQ;   // 32768
constexpr int ATS   = 40;            // padded A-tile row stride (halfs): 16+4 DWORDs

#define STR2(x) #x
#define STR(x) STR2(x)
#pragma message("compiling with clang major " STR(__clang_major__))

#if __has_builtin(__builtin_amdgcn_tensor_load_to_lds) && \
    __has_builtin(__builtin_amdgcn_s_wait_tensorcnt)
#define HAVE_TDM 1
#warning "CDNA5 TDM path ENABLED (tensor_load_to_lds + s_wait_tensorcnt)"
#else
#define HAVE_TDM 0
#warning "CDNA5 TDM path DISABLED - cooperative-load fallback in use"
#endif

#if HAVE_TDM
// Build a 2-D D# (groups 0/1 per CDNA5 ISA) and issue a
// TDM load of a tile_d0 x tile_d1 bf16 tile into LDS with 4-DWORD row padding
// (row stride in LDS becomes 20 DWORDs = ATS halfs -> conflict-free A reads).
__device__ __forceinline__ void tdm_load_tile_bf16_2d(
    unsigned lds_off, const void* gaddr,
    unsigned tensor_d0, unsigned tensor_d1,
    unsigned tile_d0, unsigned tile_d1, unsigned stride0)
{
  unsigned long long ga = (unsigned long long)(size_t)gaddr;
  u32x4 g0;
  g0.x = 1u;                                        // count=1, user descriptor
  g0.y = lds_off;                                   // lds_addr (bytes)
  g0.z = (unsigned)(ga & 0xffffffffu);              // global_addr[31:0]
  g0.w = (unsigned)((ga >> 32) & 0x01ffffffu)       // global_addr[56:32]
       | (2u << 30);                                // type = 2 ("image")
  i32x8 g1;
  g1[0] = (int)((1u << 16)                          // data_size = 1 -> 2 bytes
              | (1u << 20)                          // pad_enable
              | (3u << 22)                          // pad_interval: 16 DWORDs
              | (3u << 25));                        // pad_amount: 4 DWORDs
  g1[1] = (int)((tensor_d0 & 0xffffu) << 16);       // tensor_dim0[15:0]
  g1[2] = (int)(((tensor_d0 >> 16) & 0xffffu) | ((tensor_d1 & 0xffffu) << 16));
  g1[3] = (int)(((tensor_d1 >> 16) & 0xffffu) | ((tile_d0 & 0xffffu) << 16));
  g1[4] = (int)(tile_d1 & 0xffffu);                 // tile_dim1, tile_dim2=0
  g1[5] = (int)stride0;                             // tensor_dim0_stride[31:0]
  g1[6] = 0;
  g1[7] = 0;
  i32x4 gz = {0, 0, 0, 0};
#if __clang_major__ >= 23
  i32x8 gz8 = {0, 0, 0, 0, 0, 0, 0, 0};
  __builtin_amdgcn_tensor_load_to_lds(g0, g1, gz, gz, gz8, 0);
#else
  __builtin_amdgcn_tensor_load_to_lds(g0, g1, gz, gz, 0);
#endif
}
#endif

// ---------------------------------------------------------------------------
// Weight prep: W[K,N] f32 -> Wt[N,K] bf16 (once per launch; weights are L2-hot)
// ---------------------------------------------------------------------------
__global__ __launch_bounds__(256) void wprep_kernel(
    const float* __restrict__ W, __bf16* __restrict__ Wt, int K, int N)
{
  const int idx = blockIdx.x * 256 + threadIdx.x;   // over K*N, coalesced read
  const int k = idx / N, n = idx % N;
  Wt[(size_t)n * K + k] = (__bf16)W[idx];
}

// ---------------------------------------------------------------------------
// Embedding gather + positional add -> f32 (residual) and bf16 (GEMM A)
// ---------------------------------------------------------------------------
__global__ __launch_bounds__(256) void embed_kernel(
    const int* __restrict__ inter, const float* __restrict__ iemb,
    const float* __restrict__ pemb, float* __restrict__ xF,
    __bf16* __restrict__ xB)
{
  const int bl = blockIdx.x;
  const int d  = threadIdx.x;
  const int l  = bl & (SEQ - 1);
  const int idx = inter[bl];
  const float v = iemb[(size_t)idx * DM + d] + pemb[(size_t)l * DM + d];
  xF[(size_t)bl * DM + d] = v;
  xB[(size_t)bl * DM + d] = (__bf16)v;
}

// ---------------------------------------------------------------------------
// WMMA GEMM: C[M,N] = A_bf16[M,K] @ W (as Wt[N,K] bf16) + bias.
// Block = 8 waves -> 64x64 tile; each wave: 16x32 (2 accumulators, A reuse).
// A tile staged in LDS: double-buffered TDM pipeline (DMA of step i+1 overlaps
// WMMAs of step i), or single-buffer cooperative fallback.
// grid = (N/64, M/64).
// ---------------------------------------------------------------------------
template<bool RELU, bool STORE_F, bool STORE_B, bool STORE_BT>
__global__ __launch_bounds__(256) void gemm_bf16_wmma(
    const __bf16* __restrict__ A,
    const __bf16* __restrict__ Wt,
    const float*  __restrict__ bias,
    float*  __restrict__ Cf,
    __bf16* __restrict__ Cb,
    __bf16* __restrict__ CbT,
    int K, int N, int Mtot)
{
  __shared__ alignas(16) __bf16 atile[2][64 * ATS];

  const int tid  = threadIdx.x;
  const int lane = tid & 31;
  const int wave = tid >> 5;
  const int lr   = lane & 15;
  const int hi   = lane >> 4;
  const int m0 = blockIdx.y * 64;
  const int mt = (wave & 3) * 16;
  const int nt = (wave >> 2) * 32;
  const int col0 = blockIdx.x * 64 + nt + lr;
  const int col1 = col0 + 16;

  f32x8 acc0 = {}, acc1 = {};
  const int nsteps = K >> 5;

#if HAVE_TDM
  // -------- double-buffered TDM pipeline --------
  if (wave == 0) {
    unsigned lds0 = (unsigned)(size_t)(__attribute__((address_space(3))) void*)&atile[0][0];
    tdm_load_tile_bf16_2d(lds0, A + (size_t)m0 * K,
                          (unsigned)K, (unsigned)Mtot, 32u, 64u, (unsigned)K);
  }
  for (int ks = 0; ks < nsteps; ++ks) {
    const int k0 = ks << 5;
    if (wave == 0) __builtin_amdgcn_s_wait_tensorcnt(0);
    __syncthreads();                       // tile ks visible; tile ks-1 reads done
    if (wave == 0 && ks + 1 < nsteps) {    // overlap next DMA with this step's math
      unsigned ldsn = (unsigned)(size_t)(__attribute__((address_space(3))) void*)
                      &atile[(ks + 1) & 1][0];
      tdm_load_tile_bf16_2d(ldsn, A + (size_t)m0 * K + k0 + 32,
                            (unsigned)K, (unsigned)Mtot, 32u, 64u, (unsigned)K);
    }
    const __bf16* at = &atile[ks & 1][0];
    AFrag a;
    a.f4[0] = *(const float4*)&at[(mt + lr) * ATS + hi * 16];
    a.f4[1] = *(const float4*)&at[(mt + lr) * ATS + hi * 16 + 8];

    const __bf16* w0 = Wt + (size_t)col0 * K + k0 + hi * 16;
    const __bf16* w1 = w0 + (size_t)16 * K;
    AFrag b0, b1;
    b0.f4[0] = *(const float4*)(w0);
    b0.f4[1] = *(const float4*)(w0 + 8);
    b1.f4[0] = *(const float4*)(w1);
    b1.f4[1] = *(const float4*)(w1 + 8);
    if (k0 + 32 < K) {
      __builtin_prefetch((const void*)(w0 + 32), 0, 1);
      __builtin_prefetch((const void*)(w1 + 32), 0, 1);
    }
    acc0 = __builtin_amdgcn_wmma_f32_16x16x32_bf16(false, a.v, false, b0.v,
                                                   (short)0, acc0, false, false);
    acc1 = __builtin_amdgcn_wmma_f32_16x16x32_bf16(false, a.v, false, b1.v,
                                                   (short)0, acc1, false, false);
  }
#else
  // -------- cooperative-load fallback (single buffer) --------
  for (int ks = 0; ks < nsteps; ++ks) {
    const int k0 = ks << 5;
    {
      const int arow = tid >> 2, seg = tid & 3;
      *(float4*)&atile[0][arow * ATS + seg * 8] =
          *(const float4*)(A + (size_t)(m0 + arow) * K + k0 + seg * 8);
    }
    __syncthreads();
    AFrag a;
    a.f4[0] = *(const float4*)&atile[0][(mt + lr) * ATS + hi * 16];
    a.f4[1] = *(const float4*)&atile[0][(mt + lr) * ATS + hi * 16 + 8];

    const __bf16* w0 = Wt + (size_t)col0 * K + k0 + hi * 16;
    const __bf16* w1 = w0 + (size_t)16 * K;
    AFrag b0, b1;
    b0.f4[0] = *(const float4*)(w0);
    b0.f4[1] = *(const float4*)(w0 + 8);
    b1.f4[0] = *(const float4*)(w1);
    b1.f4[1] = *(const float4*)(w1 + 8);
    acc0 = __builtin_amdgcn_wmma_f32_16x16x32_bf16(false, a.v, false, b0.v,
                                                   (short)0, acc0, false, false);
    acc1 = __builtin_amdgcn_wmma_f32_16x16x32_bf16(false, a.v, false, b1.v,
                                                   (short)0, acc1, false, false);
    __syncthreads();
  }
#endif

  const float bv0 = bias[col0];
  const float bv1 = bias[col1];
#pragma unroll
  for (int r = 0; r < 8; ++r) {
    const int rowIdx = m0 + mt + r + 8 * hi;
    float v0 = acc0[r] + bv0;
    float v1 = acc1[r] + bv1;
    if (RELU) { v0 = fmaxf(v0, 0.f); v1 = fmaxf(v1, 0.f); }
    if (STORE_F) {
      Cf[(size_t)rowIdx * N + col0] = v0;
      Cf[(size_t)rowIdx * N + col1] = v1;
    }
    if (STORE_B) {
      Cb[(size_t)rowIdx * N + col0] = (__bf16)v0;
      Cb[(size_t)rowIdx * N + col1] = (__bf16)v1;
    }
    if (STORE_BT) {  // transposed: CbT[col][row] (for attention V operands)
      CbT[(size_t)col0 * Mtot + rowIdx] = (__bf16)v0;
      CbT[(size_t)col1 * Mtot + rowIdx] = (__bf16)v1;
    }
  }
}

// ---------------------------------------------------------------------------
// Flash attention with log-distance decay. One wave per (b, h, 16-row q tile).
// Per 32-key chunk: 2 WMMAs (Q·K^T) + online softmax + LDS relayout + 2 WMMAs (P·V).
// V comes pre-transposed (Vt[d][row]) so all operands are vector loads.
// ---------------------------------------------------------------------------
__global__ __launch_bounds__(32) void attn_kernel(
    const __bf16* __restrict__ Q, const __bf16* __restrict__ Kb,
    const __bf16* __restrict__ Vt, const unsigned char* __restrict__ mask,
    const float* __restrict__ decay_logit, __bf16* __restrict__ O)
{
  __shared__ alignas(16) __bf16 pt[16][32];

  const int lane = threadIdx.x;
  const int lr   = lane & 15;
  const int hi   = lane >> 4;
  const int q0 = blockIdx.x * 16;
  const int h  = blockIdx.y;
  const int b  = blockIdx.z;
  const float scale = 0.1767766952966369f;            // 1/sqrt(32)
  const float dec = log1pf(__expf(decay_logit[h]));   // softplus

  AFrag qa;
  {
    const __bf16* qp = Q + ((size_t)b * SEQ + q0 + lr) * DM + h * HDIM;
    qa.f4[0] = *(const float4*)(qp + hi * 16);
    qa.f4[1] = *(const float4*)(qp + hi * 16 + 8);
  }

  f32x8 o0 = {}, o1 = {};
  float mrow[8], lsum[8];
#pragma unroll
  for (int r = 0; r < 8; ++r) { mrow[r] = -3.0e38f; lsum[r] = 0.f; }

  const int kend = q0 + 16;
  for (int kb = 0; kb < kend; kb += 32) {
    AFrag bk0, bk1;
    {
      const __bf16* kp0 = Kb + ((size_t)b * SEQ + kb + lr) * DM + h * HDIM + hi * 16;
      bk0.f4[0] = *(const float4*)(kp0);
      bk0.f4[1] = *(const float4*)(kp0 + 8);
      const __bf16* kp1 = kp0 + (size_t)16 * DM;
      bk1.f4[0] = *(const float4*)(kp1);
      bk1.f4[1] = *(const float4*)(kp1 + 8);
    }
    f32x8 zero = {};
    f32x8 s0 = __builtin_amdgcn_wmma_f32_16x16x32_bf16(false, qa.v, false, bk0.v,
                                                       (short)0, zero, false, false);
    f32x8 s1 = __builtin_amdgcn_wmma_f32_16x16x32_bf16(false, qa.v, false, bk1.v,
                                                       (short)0, zero, false, false);

    const int k0i = kb + lr;
    const int k1i = kb + 16 + lr;
    const bool mk0 = mask[(size_t)b * SEQ + k0i] != 0;
    const bool mk1 = mask[(size_t)b * SEQ + k1i] != 0;

#pragma unroll
    for (int r = 0; r < 8; ++r) {
      const int qq = q0 + r + 8 * hi;
      int dd0 = qq - k0i; if (dd0 < 0) dd0 = -dd0;
      int dd1 = qq - k1i; if (dd1 < 0) dd1 = -dd1;
      float t0 = s0[r] * scale - dec * log1pf((float)dd0);
      float t1 = s1[r] * scale - dec * log1pf((float)dd1);
      if (k0i > qq || !mk0) t0 = -1e9f;
      if (k1i > qq || !mk1) t1 = -1e9f;

      float mx = fmaxf(t0, t1);
#pragma unroll
      for (int off = 1; off < 16; off <<= 1) mx = fmaxf(mx, __shfl_xor(mx, off, 32));
      const float mnew  = fmaxf(mrow[r], mx);
      const float alpha = __expf(mrow[r] - mnew);
      const float p0 = __expf(t0 - mnew);
      const float p1 = __expf(t1 - mnew);
      float ps = p0 + p1;
#pragma unroll
      for (int off = 1; off < 16; off <<= 1) ps += __shfl_xor(ps, off, 32);
      lsum[r] = lsum[r] * alpha + ps;
      mrow[r] = mnew;
      o0[r] *= alpha;
      o1[r] *= alpha;
      pt[r + 8 * hi][lr]      = (__bf16)p0;
      pt[r + 8 * hi][lr + 16] = (__bf16)p1;
    }
    __syncthreads();

    AFrag pa;
    pa.f4[0] = *(const float4*)&pt[lr][hi * 16];
    pa.f4[1] = *(const float4*)&pt[lr][hi * 16 + 8];

    const __bf16* vt0 = Vt + (size_t)(h * HDIM + lr) * ROWS + (size_t)b * SEQ + kb + hi * 16;
    const __bf16* vt1 = vt0 + (size_t)16 * ROWS;
    AFrag v0f, v1f;
    v0f.f4[0] = *(const float4*)(vt0);
    v0f.f4[1] = *(const float4*)(vt0 + 8);
    v1f.f4[0] = *(const float4*)(vt1);
    v1f.f4[1] = *(const float4*)(vt1 + 8);

    o0 = __builtin_amdgcn_wmma_f32_16x16x32_bf16(false, pa.v, false, v0f.v,
                                                 (short)0, o0, false, false);
    o1 = __builtin_amdgcn_wmma_f32_16x16x32_bf16(false, pa.v, false, v1f.v,
                                                 (short)0, o1, false, false);
    __syncthreads();
  }

#pragma unroll
  for (int r = 0; r < 8; ++r) {
    const int qq = q0 + r + 8 * hi;
    const float inv = 1.0f / lsum[r];
    __bf16* op = O + ((size_t)b * SEQ + qq) * DM + h * HDIM;
    op[lr]      = (__bf16)(o0[r] * inv);
    op[16 + lr] = (__bf16)(o1[r] * inv);
  }
}

// ---------------------------------------------------------------------------
// Fused residual add + LayerNorm over D=256. One wave per row, 8 rows/block.
// ---------------------------------------------------------------------------
__global__ __launch_bounds__(256) void add_ln_kernel(
    const float* __restrict__ X, const float* __restrict__ Y,
    const float* __restrict__ G, const float* __restrict__ Bt,
    float* __restrict__ outF, __bf16* __restrict__ outB)
{
  const int lane = threadIdx.x & 31;
  const int wave = threadIdx.x >> 5;
  const size_t row = (size_t)blockIdx.x * 8 + wave;
  const float* xr = X + row * DM;
  const float* yr = Y + row * DM;
  float v[8];
  float s = 0.f;
#pragma unroll
  for (int i = 0; i < 8; ++i) { v[i] = xr[lane + i * 32] + yr[lane + i * 32]; s += v[i]; }
#pragma unroll
  for (int off = 1; off < 32; off <<= 1) s += __shfl_xor(s, off, 32);
  const float mu = s * (1.0f / DM);
  float q = 0.f;
#pragma unroll
  for (int i = 0; i < 8; ++i) { const float d = v[i] - mu; q += d * d; }
#pragma unroll
  for (int off = 1; off < 32; off <<= 1) q += __shfl_xor(q, off, 32);
  const float rs = rsqrtf(q * (1.0f / DM) + 1e-5f);
#pragma unroll
  for (int i = 0; i < 8; ++i) {
    const int d = lane + i * 32;
    const float o = (v[i] - mu) * rs * G[d] + Bt[d];
    outF[row * DM + d] = o;
    if (outB) outB[row * DM + d] = (__bf16)o;
  }
}

// ---------------------------------------------------------------------------
// Final: out[row] = x2·Wout[0:256] + skill_emb[ts]·Wout[256:512] + bout
// ---------------------------------------------------------------------------
__global__ __launch_bounds__(256) void out_kernel(
    const float* __restrict__ X2, const int* __restrict__ tsk,
    const float* __restrict__ semb, const float* __restrict__ Wout,
    const float* __restrict__ bout, float* __restrict__ out)
{
  const int lane = threadIdx.x & 31;
  const int wave = threadIdx.x >> 5;
  const size_t row = (size_t)blockIdx.x * 8 + wave;
  const float* xr = X2 + row * DM;
  const float* sr = semb + (size_t)tsk[row] * DM;
  float s = 0.f;
#pragma unroll
  for (int i = 0; i < 8; ++i) {
    const int d = lane + i * 32;
    s += xr[d] * Wout[d] + sr[d] * Wout[DM + d];
  }
#pragma unroll
  for (int off = 1; off < 32; off <<= 1) s += __shfl_xor(s, off, 32);
  if (lane == 0) out[row] = s + bout[0];
}

// ---------------------------------------------------------------------------
extern "C" void kernel_launch(void* const* d_in, const int* in_sizes, int n_in,
                              void* d_out, int out_size, void* d_ws, size_t ws_size,
                              hipStream_t stream)
{
  (void)in_sizes; (void)n_in; (void)out_size; (void)ws_size;

  const int*   inter = (const int*)d_in[0];
  const int*   tsk   = (const int*)d_in[1];
  const unsigned char* mask = (const unsigned char*)d_in[2];
  const float* iemb = (const float*)d_in[3];
  const float* pemb = (const float*)d_in[4];
  const float* Wq = (const float*)d_in[5];  const float* bq = (const float*)d_in[6];
  const float* Wk = (const float*)d_in[7];  const float* bk = (const float*)d_in[8];
  const float* Wv = (const float*)d_in[9];  const float* bv = (const float*)d_in[10];
  const float* Wo = (const float*)d_in[11]; const float* bo = (const float*)d_in[12];
  const float* dlog = (const float*)d_in[13];
  const float* g1 = (const float*)d_in[14]; const float* be1 = (const float*)d_in[15];
  const float* g2 = (const float*)d_in[16]; const float* be2 = (const float*)d_in[17];
  const float* W1 = (const float*)d_in[18]; const float* b1 = (const float*)d_in[19];
  const float* W2 = (const float*)d_in[20]; const float* b2 = (const float*)d_in[21];
  const float* semb = (const float*)d_in[22];
  const float* Wout = (const float*)d_in[23]; const float* bout = (const float*)d_in[24];
  float* out = (float*)d_out;

  // Workspace: 3x f32 [ROWS,DM] (96MB) + 4x bf16 [ROWS,DM] (64MB) + Wt pool (~1MB)
  char* ws = (char*)d_ws;
  const size_t F  = (size_t)ROWS * DM * sizeof(float);
  const size_t Hh = (size_t)ROWS * DM * sizeof(__bf16);
  float*  xF  = (float*)(ws);            // x (embed), reused for x2
  float*  yF  = (float*)(ws + F);        // x_attn, later ffn_out
  float*  x1F = (float*)(ws + 2 * F);    // post-LN1
  __bf16* xB  = (__bf16*)(ws + 3 * F);
  __bf16* qB  = (__bf16*)(ws + 3 * F + Hh);
  __bf16* kB  = (__bf16*)(ws + 3 * F + 2 * Hh);
  __bf16* vT  = (__bf16*)(ws + 3 * F + 3 * Hh);  // transposed V [DM][ROWS]
  __bf16* oB  = xB;                      // xB dead after QKV GEMMs
  __bf16* x1B = qB;                      // qB dead after attention
  __bf16* hB  = kB;                      // 32MB spanning kB..vT, dead after attention
  float*  x2F = xF;
  char* wtp = ws + 3 * F + 4 * Hh;
  const size_t WSQ = (size_t)DM * DM * sizeof(__bf16);   // 128KB
  __bf16* WqT = (__bf16*)(wtp);
  __bf16* WkT = (__bf16*)(wtp + WSQ);
  __bf16* WvT = (__bf16*)(wtp + 2 * WSQ);
  __bf16* WoT = (__bf16*)(wtp + 3 * WSQ);
  __bf16* W1T = (__bf16*)(wtp + 4 * WSQ);                // 256KB
  __bf16* W2T = (__bf16*)(wtp + 6 * WSQ);                // 256KB

  const dim3 blk256(256);
  wprep_kernel<<<dim3(DM * DM / 256), blk256, 0, stream>>>(Wq, WqT, DM, DM);
  wprep_kernel<<<dim3(DM * DM / 256), blk256, 0, stream>>>(Wk, WkT, DM, DM);
  wprep_kernel<<<dim3(DM * DM / 256), blk256, 0, stream>>>(Wv, WvT, DM, DM);
  wprep_kernel<<<dim3(DM * DM / 256), blk256, 0, stream>>>(Wo, WoT, DM, DM);
  wprep_kernel<<<dim3(DM * DFF / 256), blk256, 0, stream>>>(W1, W1T, DM, DFF);
  wprep_kernel<<<dim3(DFF * DM / 256), blk256, 0, stream>>>(W2, W2T, DFF, DM);

  embed_kernel<<<dim3(ROWS), blk256, 0, stream>>>(inter, iemb, pemb, xF, xB);

  const dim3 gD(DM / 64, ROWS / 64);
  gemm_bf16_wmma<false, false, true, false><<<gD, blk256, 0, stream>>>(
      xB, WqT, bq, nullptr, qB, nullptr, DM, DM, ROWS);
  gemm_bf16_wmma<false, false, true, false><<<gD, blk256, 0, stream>>>(
      xB, WkT, bk, nullptr, kB, nullptr, DM, DM, ROWS);
  gemm_bf16_wmma<false, false, false, true><<<gD, blk256, 0, stream>>>(
      xB, WvT, bv, nullptr, nullptr, vT, DM, DM, ROWS);

  attn_kernel<<<dim3(SEQ / 16, NH, BATCH), dim3(32), 0, stream>>>(
      qB, kB, vT, mask, dlog, oB);

  gemm_bf16_wmma<false, true, false, false><<<gD, blk256, 0, stream>>>(
      oB, WoT, bo, yF, nullptr, nullptr, DM, DM, ROWS);

  add_ln_kernel<<<dim3(ROWS / 8), blk256, 0, stream>>>(xF, yF, g1, be1, x1F, x1B);

  gemm_bf16_wmma<true, false, true, false><<<dim3(DFF / 64, ROWS / 64), blk256, 0, stream>>>(
      x1B, W1T, b1, nullptr, hB, nullptr, DM, DFF, ROWS);
  gemm_bf16_wmma<false, true, false, false><<<gD, blk256, 0, stream>>>(
      hB, W2T, b2, yF, nullptr, nullptr, DFF, DM, ROWS);

  add_ln_kernel<<<dim3(ROWS / 8), blk256, 0, stream>>>(x1F, yF, g2, be2, x2F, nullptr);

  out_kernel<<<dim3(ROWS / 8), blk256, 0, stream>>>(x2F, tsk, semb, Wout, bout, out);
}
